// LittleBitLinear_15702400434326
// MI455X (gfx1250) — matrix operand order repaired and checked by reading, not compiled, not run on hardware
//
#include <hip/hip_runtime.h>
#include <cstdint>
#include <cstddef>

typedef __attribute__((ext_vector_type(16))) __bf16    v16bf;
typedef __attribute__((ext_vector_type(8)))  __bf16    v8bf;
typedef __attribute__((ext_vector_type(8)))  float     v8f;
typedef __attribute__((ext_vector_type(4)))  uint32_t  v4u;
typedef __attribute__((ext_vector_type(8)))  uint32_t  v8u;

#define IN_DIM   4096
#define OUT_DIM  4096
#define SPLIT    1024
#define NROWS    8192
#define LDSP     40   // padded LDS row stride in bf16 elems: 80B rows (64B data + 16B pad)

// ---------------- prep kernels: fold diagonal scales into sign matrices ----------------

__global__ void prep_w1_kernel(const float* __restrict__ V, const float* __restrict__ v2,
                               const float* __restrict__ V_R, const float* __restrict__ v2_R,
                               __bf16* __restrict__ W1a, __bf16* __restrict__ W1b) {
    int idx = blockIdx.x * blockDim.x + threadIdx.x;   // over SPLIT*IN_DIM
    int k = idx & (IN_DIM - 1);
    float sa = (V[idx]   >= 0.f) ? 1.f : -1.f;
    float sb = (V_R[idx] >= 0.f) ? 1.f : -1.f;
    W1a[idx] = (__bf16)(sa * v2[k]);
    W1b[idx] = (__bf16)(sb * v2_R[k]);
}

__global__ void prep_w2_kernel(const float* __restrict__ U, const float* __restrict__ v1,
                               const float* __restrict__ u2, const float* __restrict__ u1,
                               const float* __restrict__ U_R, const float* __restrict__ v1_R,
                               const float* __restrict__ u2_R, const float* __restrict__ u1_R,
                               __bf16* __restrict__ W2a, __bf16* __restrict__ W2b) {
    int idx = blockIdx.x * blockDim.x + threadIdx.x;   // over OUT_DIM*SPLIT
    int s = idx & (SPLIT - 1);
    int o = idx >> 10;
    float sa = (U[idx]   >= 0.f) ? 1.f : -1.f;
    float sb = (U_R[idx] >= 0.f) ? 1.f : -1.f;
    W2a[idx] = (__bf16)(sa * v1[s]   * u2[s]   * u1[o]);
    W2b[idx] = (__bf16)(sb * v1_R[s] * u2_R[s] * u1_R[o]);
}

// ---------------- Tensor Data Mover: 2D tile load (128 rows x 32 bf16) -> padded LDS ----------------
// D# per CDNA5 ISA ch.8: group0 = {flags, lds_addr, global_addr}, group1 = dims/strides/pad.
// pad_enable: insert 4 DWORDs (16B) after every 16 DWORDs (64B) stored -> 80B LDS row = LDSP elems.

__device__ __forceinline__ void tdm_load_tile_128x32(const void* gsrc, uint32_t ldsAddr,
                                                     uint32_t rowStrideElems) {
    uint64_t ga = (uint64_t)(uintptr_t)gsrc;
    v4u g0;
    g0[0] = 1u;                                                  // count=1 (valid), user mode
    g0[1] = ldsAddr;                                             // lds_addr (bytes)
    g0[2] = (uint32_t)ga;                                        // global_addr[31:0]
    g0[3] = ((uint32_t)(ga >> 32) & 0x1FFFFFFu) | (2u << 30);    // global_addr[56:32] | type=2
    v8u g1;
    g1[0] = (1u << 16)      // data_size = 1 -> 2 bytes
          | (1u << 20)      // pad_enable
          | (3u << 22)      // pad_interval: 16 DWORDs between pads
          | (3u << 25);     // pad_amount: 4 DWORDs (16B)
    g1[1] = (32u << 16);                 // tensor_dim0 = 32 (lo16)
    g1[2] = (128u << 16);                // tensor_dim0 hi = 0 | tensor_dim1 = 128 (lo16)
    g1[3] = (32u << 16);                 // tensor_dim1 hi = 0 | tile_dim0 = 32
    g1[4] = 128u;                        // tile_dim1 = 128 | tile_dim2 = 0
    g1[5] = rowStrideElems;              // tensor_dim0_stride[31:0] (elements)
    g1[6] = 0u;                          // stride hi | tensor_dim1_stride lo (unused, 2D)
    g1[7] = 0u;
    v4u gz = {0u, 0u, 0u, 0u};
    asm volatile("tensor_load_to_lds %0, %1, %2, %3"
                 :: "s"(g0), "s"(g1), "s"(gz), "s"(gz)
                 : "memory");
}

__device__ __forceinline__ void tdm_wait0() {
    asm volatile("s_wait_tensorcnt 0x0" ::: "memory");
}

// ---------------- fragment helpers (layouts per ISA 7.12.2, wave32) ----------------

__device__ __forceinline__ v16bf load_a_frag(const __bf16* sA, int rowBase, int lrow, int lhi) {
    const __bf16* p = sA + (rowBase + lrow) * LDSP + lhi * 8;
    v8bf lo = *(const v8bf*)(p);
    v8bf hi = *(const v8bf*)(p + 16);
    return __builtin_shufflevector(lo, hi, 0,1,2,3,4,5,6,7,8,9,10,11,12,13,14,15);
}

__device__ __forceinline__ v16bf load_b_frag(const __bf16* sB, int colBase, int lrow, int lhi) {
    const __bf16* p = sB + (colBase + lrow) * LDSP + lhi * 8;
    v8bf lo = *(const v8bf*)(p);
    v8bf hi = *(const v8bf*)(p + 16);
    return __builtin_shufflevector(lo, hi, 0,1,2,3,4,5,6,7,8,9,10,11,12,13,14,15);
}

// ---------------- GEMM1: h = bf16(x) @ W1^T ----------------
// M=8192, N=SPLIT, K=IN.  Block 128x128x32, 256 threads / 8 waves, wave tile 32x64.
// Double-buffered LDS; B tile via TDM, A tile via f32 loads + VALU cvt (register-prefetched).
// branch folded into fast-varying blockIdx.y for L2 reuse of x.

__global__ __launch_bounds__(256)
void gemm1_kernel(const float* __restrict__ x,
                  const __bf16* __restrict__ W1a, const __bf16* __restrict__ W1b,
                  __bf16* __restrict__ h1, __bf16* __restrict__ h2) {
    __shared__ __bf16 sA[2][128 * LDSP];
    __shared__ __bf16 sB[2][128 * LDSP];

    const int branch = blockIdx.y & 1;
    const int nBase  = (blockIdx.y >> 1) * 128;
    const __bf16* __restrict__ B = branch ? W1b : W1a;
    __bf16* __restrict__ H       = branch ? h2  : h1;

    const int tid  = threadIdx.x;
    const int rowBase = blockIdx.x * 128;
    const int wave = tid >> 5, lane = tid & 31;
    const int wm = (wave & 3) * 32;
    const int wn = (wave >> 2) * 64;
    const int lrow = lane & 15, lhi = lane >> 4;
    const int lr = tid >> 1, lc = (tid & 1) * 16;

    const uint32_t sB0 = (uint32_t)(uintptr_t)&sB[0][0];
    const uint32_t sB1 = (uint32_t)(uintptr_t)&sB[1][0];
    const __bf16* gBbase = B + (size_t)nBase * IN_DIM;
    const float*  gAbase = x + (size_t)(rowBase + lr) * IN_DIM + lc;

    v8f acc[2][4] = {};

    // ---- prologue: tile 0 into buffer 0 ----
    if (wave == 0) tdm_load_tile_128x32(gBbase, sB0, IN_DIM);
    {
        const float* gA = gAbase;
        float4 f0 = ((const float4*)gA)[0];
        float4 f1 = ((const float4*)gA)[1];
        float4 f2 = ((const float4*)gA)[2];
        float4 f3 = ((const float4*)gA)[3];
        v8bf p0, p1;
        p0[0]=(__bf16)f0.x; p0[1]=(__bf16)f0.y; p0[2]=(__bf16)f0.z; p0[3]=(__bf16)f0.w;
        p0[4]=(__bf16)f1.x; p0[5]=(__bf16)f1.y; p0[6]=(__bf16)f1.z; p0[7]=(__bf16)f1.w;
        p1[0]=(__bf16)f2.x; p1[1]=(__bf16)f2.y; p1[2]=(__bf16)f2.z; p1[3]=(__bf16)f2.w;
        p1[4]=(__bf16)f3.x; p1[5]=(__bf16)f3.y; p1[6]=(__bf16)f3.z; p1[7]=(__bf16)f3.w;
        *(v8bf*)&sA[0][lr * LDSP + lc]     = p0;
        *(v8bf*)&sA[0][lr * LDSP + lc + 8] = p1;
    }
    if (wave == 0) tdm_wait0();
    __syncthreads();

    int cur = 0;
    const int NK = IN_DIM / 32;
    for (int kt = 0; kt < NK; ++kt) {
        const int nxt = cur ^ 1;
        const bool have_next = (kt + 1 < NK);

        // issue next B tile via TDM early (overlaps with compute below)
        if (have_next && wave == 0)
            tdm_load_tile_128x32(gBbase + (kt + 1) * 32, nxt ? sB1 : sB0, IN_DIM);

        // prefetch next A tile into registers
        float4 f0, f1, f2, f3;
        if (have_next) {
            const float* gA = gAbase + (kt + 1) * 32;
            f0 = ((const float4*)gA)[0];
            f1 = ((const float4*)gA)[1];
            f2 = ((const float4*)gA)[2];
            f3 = ((const float4*)gA)[3];
        }

        // compute from current buffers
        v16bf af[2], bfr[4];
#pragma unroll
        for (int mi = 0; mi < 2; ++mi) af[mi]  = load_a_frag(&sA[cur][0], wm + mi * 16, lrow, lhi);
#pragma unroll
        for (int ni = 0; ni < 4; ++ni) bfr[ni] = load_b_frag(&sB[cur][0], wn + ni * 16, lrow, lhi);
#pragma unroll
        for (int mi = 0; mi < 2; ++mi)
#pragma unroll
            for (int ni = 0; ni < 4; ++ni)
                acc[mi][ni] = __builtin_amdgcn_wmma_f32_16x16x32_bf16(
                    false, af[mi], false, bfr[ni], (short)0, acc[mi][ni], false, false);

        // stage next A tile into the other buffer
        if (have_next) {
            v8bf p0, p1;
            p0[0]=(__bf16)f0.x; p0[1]=(__bf16)f0.y; p0[2]=(__bf16)f0.z; p0[3]=(__bf16)f0.w;
            p0[4]=(__bf16)f1.x; p0[5]=(__bf16)f1.y; p0[6]=(__bf16)f1.z; p0[7]=(__bf16)f1.w;
            p1[0]=(__bf16)f2.x; p1[1]=(__bf16)f2.y; p1[2]=(__bf16)f2.z; p1[3]=(__bf16)f2.w;
            p1[4]=(__bf16)f3.x; p1[5]=(__bf16)f3.y; p1[6]=(__bf16)f3.z; p1[7]=(__bf16)f3.w;
            *(v8bf*)&sA[nxt][lr * LDSP + lc]     = p0;
            *(v8bf*)&sA[nxt][lr * LDSP + lc + 8] = p1;
        }

        if (wave == 0) tdm_wait0();
        __syncthreads();
        cur = nxt;
    }

    // ---- epilogue: C VGPR r -> (M = r + 8*lhi, N = lrow) ----
#pragma unroll
    for (int mi = 0; mi < 2; ++mi)
#pragma unroll
        for (int ni = 0; ni < 4; ++ni)
#pragma unroll
            for (int r = 0; r < 8; ++r) {
                int row = rowBase + wm + mi * 16 + r + lhi * 8;
                int col = nBase + wn + ni * 16 + lrow;
                H[(size_t)row * SPLIT + col] = (__bf16)acc[mi][ni][r];
            }
}

// ---------------- GEMM2: out = h1 @ W2a^T + h2 @ W2b^T + bias ----------------
// Both A and B tiles via TDM, double-buffered; flattened 64-step K loop over the two branches.

__global__ __launch_bounds__(256)
void gemm2_kernel(const __bf16* __restrict__ h1, const __bf16* __restrict__ h2,
                  const __bf16* __restrict__ W2a, const __bf16* __restrict__ W2b,
                  const float* __restrict__ bias, float* __restrict__ out) {
    __shared__ __bf16 sA[2][128 * LDSP];
    __shared__ __bf16 sB[2][128 * LDSP];

    const int tid  = threadIdx.x;
    const int rowBase = blockIdx.x * 128;
    const int nBase   = blockIdx.y * 128;
    const int wave = tid >> 5, lane = tid & 31;
    const int wm = (wave & 3) * 32;
    const int wn = (wave >> 2) * 64;
    const int lrow = lane & 15, lhi = lane >> 4;

    const uint32_t sAaddr[2] = { (uint32_t)(uintptr_t)&sA[0][0], (uint32_t)(uintptr_t)&sA[1][0] };
    const uint32_t sBaddr[2] = { (uint32_t)(uintptr_t)&sB[0][0], (uint32_t)(uintptr_t)&sB[1][0] };
    const __bf16* Abase[2] = { h1 + (size_t)rowBase * SPLIT, h2 + (size_t)rowBase * SPLIT };
    const __bf16* Bbase[2] = { W2a + (size_t)nBase * SPLIT,  W2b + (size_t)nBase * SPLIT };

    v8f acc[2][4] = {};

    // prologue: tile 0
    if (wave == 0) {
        tdm_load_tile_128x32(Abase[0], sAaddr[0], SPLIT);
        tdm_load_tile_128x32(Bbase[0], sBaddr[0], SPLIT);
        tdm_wait0();
    }
    __syncthreads();

    int cur = 0;
    const int NK = 2 * (SPLIT / 32);   // 64 steps: branch = kt>>5, kk = kt&31
    for (int kt = 0; kt < NK; ++kt) {
        const int nxt = cur ^ 1;
        const bool have_next = (kt + 1 < NK);

        if (have_next && wave == 0) {
            const int kn = kt + 1;
            tdm_load_tile_128x32(Abase[kn >> 5] + (kn & 31) * 32, sAaddr[nxt], SPLIT);
            tdm_load_tile_128x32(Bbase[kn >> 5] + (kn & 31) * 32, sBaddr[nxt], SPLIT);
        }

        v16bf af[2], bfr[4];
#pragma unroll
        for (int mi = 0; mi < 2; ++mi) af[mi]  = load_a_frag(&sA[cur][0], wm + mi * 16, lrow, lhi);
#pragma unroll
        for (int ni = 0; ni < 4; ++ni) bfr[ni] = load_b_frag(&sB[cur][0], wn + ni * 16, lrow, lhi);
#pragma unroll
        for (int mi = 0; mi < 2; ++mi)
#pragma unroll
            for (int ni = 0; ni < 4; ++ni)
                acc[mi][ni] = __builtin_amdgcn_wmma_f32_16x16x32_bf16(
                    false, af[mi], false, bfr[ni], (short)0, acc[mi][ni], false, false);

        if (wave == 0) tdm_wait0();
        __syncthreads();
        cur = nxt;
    }

    // epilogue: fp32 out + bias
#pragma unroll
    for (int mi = 0; mi < 2; ++mi)
#pragma unroll
        for (int ni = 0; ni < 4; ++ni)
#pragma unroll
            for (int r = 0; r < 8; ++r) {
                int row = rowBase + wm + mi * 16 + r + lhi * 8;
                int col = nBase + wn + ni * 16 + lrow;
                out[(size_t)row * OUT_DIM + col] = acc[mi][ni][r] + bias[col];
            }
}

// ---------------- launch ----------------

extern "C" void kernel_launch(void* const* d_in, const int* in_sizes, int n_in,
                              void* d_out, int out_size, void* d_ws, size_t ws_size,
                              hipStream_t stream) {
    const float* x    = (const float*)d_in[0];
    const float* V    = (const float*)d_in[1];
    const float* U    = (const float*)d_in[2];
    const float* v2   = (const float*)d_in[3];
    const float* v1   = (const float*)d_in[4];
    const float* u2   = (const float*)d_in[5];
    const float* u1   = (const float*)d_in[6];
    const float* V_R  = (const float*)d_in[7];
    const float* U_R  = (const float*)d_in[8];
    const float* v2_R = (const float*)d_in[9];
    const float* v1_R = (const float*)d_in[10];
    const float* u2_R = (const float*)d_in[11];
    const float* u1_R = (const float*)d_in[12];
    const float* bias = (const float*)d_in[13];
    float* out = (float*)d_out;

    char* ws = (char*)d_ws;
    const size_t MB = 1024u * 1024u;
    __bf16* W1a = (__bf16*)(ws + 0 * MB);    // 8 MiB  [SPLIT][IN]
    __bf16* W1b = (__bf16*)(ws + 8 * MB);    // 8 MiB
    __bf16* W2a = (__bf16*)(ws + 16 * MB);   // 8 MiB  [OUT][SPLIT]
    __bf16* W2b = (__bf16*)(ws + 24 * MB);   // 8 MiB
    __bf16* h1  = (__bf16*)(ws + 32 * MB);   // 16 MiB [NROWS][SPLIT]
    __bf16* h2  = (__bf16*)(ws + 48 * MB);   // 16 MiB

    prep_w1_kernel<<<(SPLIT * IN_DIM) / 256, 256, 0, stream>>>(V, v2, V_R, v2_R, W1a, W1b);
    prep_w2_kernel<<<(OUT_DIM * SPLIT) / 256, 256, 0, stream>>>(U, v1, u2, u1,
                                                                U_R, v1_R, u2_R, u1_R, W2a, W2b);
    gemm1_kernel<<<dim3(NROWS / 128, (SPLIT / 128) * 2), 256, 0, stream>>>(x, W1a, W1b, h1, h2);
    gemm2_kernel<<<dim3(NROWS / 128, OUT_DIM / 128), 256, 0, stream>>>(h1, h2, W2a, W2b, bias, out);
}
